// ResidualSimVQ_45148696216567
// MI455X (gfx1250) — matrix operand order repaired and checked by reading, not compiled, not run on hardware
//
#include <hip/hip_runtime.h>

#define NROW   32768          // 8 * 4096 rows
#define DIMD   256            // feature dim
#define KCB    4096           // codebook size
#define NQ     4              // quantizers
#define NELEM  (NROW * DIMD)  // 8388608

#define PITCH_E 264           // LDS row pitch in bf16 elements (528 B = 132 DW)
#define NTILE   (KCB / 16)    // 256 column tiles

typedef __attribute__((ext_vector_type(16))) __bf16 v16bf;
typedef __attribute__((ext_vector_type(8)))  float  v8f;

union ABFrag {
    v16bf v;
    struct { uint4 lo, hi; } q;
};

__device__ __forceinline__ unsigned short f2bf(float f) {
    unsigned int u = __float_as_uint(f);
    u += 0x7FFFu + ((u >> 16) & 1u);   // round-to-nearest-even
    return (unsigned short)(u >> 16);
}

// ---------------------------------------------------------------------------
// init: residual = x, res_bf = bf16(x), qout = 0, loss accumulators = 0
// ---------------------------------------------------------------------------
__global__ __launch_bounds__(256) void init_kernel(
    const float* __restrict__ x, float* __restrict__ residual,
    unsigned short* __restrict__ res_bf, float* __restrict__ qout,
    float* __restrict__ lacc)
{
    if (blockIdx.x == 0 && threadIdx.x < NQ) lacc[threadIdx.x] = 0.0f;
    for (int i = blockIdx.x * blockDim.x + threadIdx.x; i < NELEM;
         i += gridDim.x * blockDim.x) {
        float v = x[i];
        residual[i] = v;
        res_bf[i]   = f2bf(v);
        qout[i]     = 0.0f;
    }
}

// codebooks f32 -> bf16
__global__ __launch_bounds__(256) void cbconv_kernel(
    const float* __restrict__ cb, unsigned short* __restrict__ cb_bf)
{
    const int total = NQ * KCB * DIMD;
    for (int i = blockIdx.x * blockDim.x + threadIdx.x; i < total;
         i += gridDim.x * blockDim.x)
        cb_bf[i] = f2bf(cb[i]);
}

// per-code squared norms (one wave per code)
__global__ __launch_bounds__(256) void cnorm_kernel(
    const float* __restrict__ cb, float* __restrict__ cnorm)
{
    int lane = threadIdx.x & 31;
    int wave = threadIdx.x >> 5;
    int code = blockIdx.x * 8 + wave;          // grid = (NQ*KCB)/8 blocks
    if (code >= NQ * KCB) return;
    const float* row = cb + (size_t)code * DIMD;
    const float4* r4 = (const float4*)(row) + lane * 2;   // 8 floats per lane
    float4 a = r4[0], b = r4[1];
    float s = a.x*a.x + a.y*a.y + a.z*a.z + a.w*a.w
            + b.x*b.x + b.y*b.y + b.z*b.z + b.w*b.w;
    #pragma unroll
    for (int m = 16; m >= 1; m >>= 1) s += __shfl_xor(s, m, 32);
    if (lane == 0) cnorm[code] = s;
}

// ---------------------------------------------------------------------------
// Fused distance GEMM + argmin.
// Block = 256 threads (8 waves); each wave owns 32 rows (two 16-row A tiles
// held in VGPRs). The block marches over all 256 codebook column tiles; each
// tile (16 codes x 256 dims, bf16 = 8KB) is staged into LDS once per block
// via async global->LDS copies, double-buffered and prefetched one tile ahead.
// dist(row, code) = cnorm[code] - 2 * <a_row, c_code>   (argmin-equivalent)
// ---------------------------------------------------------------------------
__global__ __launch_bounds__(256) void argmin_kernel(
    const unsigned short* __restrict__ res_bf,   // [NROW, 256] bf16
    const unsigned short* __restrict__ cb_bf,    // [4096, 256] bf16 (layer)
    const float* __restrict__ cnorm,             // [4096]
    int* __restrict__ idx_out)                   // [NROW]
{
    __shared__ unsigned short ldsB[2][16 * PITCH_E];   // 2 x 8448 B

    const int tid  = threadIdx.x;
    const int lane = tid & 31;
    const int wave = tid >> 5;
    const int row0 = (blockIdx.x * 8 + wave) * 32;     // 32 rows per wave
    const int half = lane >> 4;        // 0: lanes 0-15, 1: lanes 16-31
    const int l16  = lane & 15;
    const int akoff = half * 8;        // A: K interleaved by 8 between halves
    const int boff  = half * 8;        // B: K interleaved by 8 between halves

    // Async-copy slice for this thread: row r (16 rows), 32-byte chunk c (16).
    const int cp_r = tid >> 4;
    const int cp_c = tid & 15;
    const unsigned long long cp_src0 =
        (unsigned long long)(cb_bf + (size_t)cp_r * DIMD + cp_c * 16);
    const unsigned int cp_dst0 =
        (unsigned int)(unsigned long long)&ldsB[0][cp_r * PITCH_E + cp_c * 16];
    const unsigned int cp_dst1 =
        (unsigned int)(unsigned long long)&ldsB[1][cp_r * PITCH_E + cp_c * 16];

    #define ISSUE_TILE(J, DST)                                              \
    do {                                                                    \
        unsigned long long ga = cp_src0 + (unsigned long long)(J) * (16 * DIMD * 2); \
        asm volatile("global_load_async_to_lds_b128 %0, %1, off\n\t"        \
                     "global_load_async_to_lds_b128 %0, %1, off offset:16"  \
                     :: "v"(DST), "v"(ga) : "memory");                      \
    } while (0)

    // Load this wave's 32x256 A tile into registers (2 x 8 K-tiles of 32).
    ABFrag a0[8], a1[8];
    {
        const unsigned short* ar0 = res_bf + (size_t)(row0 + l16) * DIMD;
        const unsigned short* ar1 = ar0 + 16 * DIMD;
        #pragma unroll
        for (int t = 0; t < 8; ++t) {
            const int k0 = t * 32;
            a0[t].q.lo = *(const uint4*)(ar0 + k0 + akoff);
            a0[t].q.hi = *(const uint4*)(ar0 + k0 + 16 + akoff);
            a1[t].q.lo = *(const uint4*)(ar1 + k0 + akoff);
            a1[t].q.hi = *(const uint4*)(ar1 + k0 + 16 + akoff);
        }
    }

    float minv0[8], minv1[8];
    int   mini0[8], mini1[8];
    #pragma unroll
    for (int r = 0; r < 8; ++r) {
        minv0[r] = 3.4e38f; mini0[r] = 0;
        minv1[r] = 3.4e38f; mini1[r] = 0;
    }

    ISSUE_TILE(0, cp_dst0);                       // prologue prefetch

    for (int j = 0; j < NTILE; ++j) {
        const unsigned short* buf = ldsB[j & 1];
        __syncthreads();   // readers of the buffer we are about to refill are done
        if (j + 1 < NTILE) {
            ISSUE_TILE(j + 1, (j & 1) ? cp_dst0 : cp_dst1);
            asm volatile("s_wait_asynccnt 0x2" ::: "memory");  // tile j landed
        } else {
            asm volatile("s_wait_asynccnt 0x0" ::: "memory");
        }
        __syncthreads();   // tile j visible to all waves

        const int col = j * 16 + l16;
        const float cn = cnorm[col];

        v8f c0a = {}, c0b = {}, c1a = {}, c1b = {};
        const unsigned short* brow = buf + l16 * PITCH_E + boff;
        #pragma unroll
        for (int t = 0; t < 4; ++t) {
            ABFrag bE, bO;
            const unsigned short* p = brow + t * 64;      // K = t*64 (+half*8)
            bE.q.lo = *(const uint4*)(p);
            bE.q.hi = *(const uint4*)(p + 16);
            bO.q.lo = *(const uint4*)(p + 32);            // K = t*64+32
            bO.q.hi = *(const uint4*)(p + 48);
            c0a = __builtin_amdgcn_wmma_f32_16x16x32_bf16(
                      false, a0[2*t].v,   false, bE.v, (short)0, c0a, false, false);
            c1a = __builtin_amdgcn_wmma_f32_16x16x32_bf16(
                      false, a1[2*t].v,   false, bE.v, (short)0, c1a, false, false);
            c0b = __builtin_amdgcn_wmma_f32_16x16x32_bf16(
                      false, a0[2*t+1].v, false, bO.v, (short)0, c0b, false, false);
            c1b = __builtin_amdgcn_wmma_f32_16x16x32_bf16(
                      false, a1[2*t+1].v, false, bO.v, (short)0, c1b, false, false);
        }
        #pragma unroll
        for (int r = 0; r < 8; ++r) {
            float d0 = cn - 2.0f * (c0a[r] + c0b[r]);
            if (d0 < minv0[r]) { minv0[r] = d0; mini0[r] = col; }
            float d1 = cn - 2.0f * (c1a[r] + c1b[r]);
            if (d1 < minv1[r]) { minv1[r] = d1; mini1[r] = col; }
        }
    }
    #undef ISSUE_TILE

    // Reduce across the 16 lanes of each half-wave.
    // C layout: VGPR r -> row (base+r) in lanes 0-15, row (base+8+r) in 16-31.
    #pragma unroll
    for (int r = 0; r < 8; ++r) {
        float mv0 = minv0[r]; int mi0 = mini0[r];
        float mv1 = minv1[r]; int mi1 = mini1[r];
        #pragma unroll
        for (int s = 1; s < 16; s <<= 1) {
            float ov = __shfl_xor(mv0, s, 32);
            int   oi = __shfl_xor(mi0, s, 32);
            if (ov < mv0 || (ov == mv0 && oi < mi0)) { mv0 = ov; mi0 = oi; }
            ov = __shfl_xor(mv1, s, 32);
            oi = __shfl_xor(mi1, s, 32);
            if (ov < mv1 || (ov == mv1 && oi < mi1)) { mv1 = ov; mi1 = oi; }
        }
        if (l16 == 0) {
            idx_out[row0 + half * 8 + r]      = mi0;
            idx_out[row0 + 16 + half * 8 + r] = mi1;
        }
    }
}

// ---------------------------------------------------------------------------
// residual -= q; qout += q; res_bf = bf16(residual); loss += residual^2
// ---------------------------------------------------------------------------
__global__ __launch_bounds__(256) void update_kernel(
    float* __restrict__ residual, unsigned short* __restrict__ res_bf,
    const float* __restrict__ cb,            // layer's f32 codebook
    const int* __restrict__ idx,
    float* __restrict__ qout, float* __restrict__ lacc)
{
    float s = 0.0f;
    for (int i = blockIdx.x * blockDim.x + threadIdx.x; i < NELEM;
         i += gridDim.x * blockDim.x) {
        const int n = i >> 8;
        const int d = i & 255;
        const float q = cb[(size_t)idx[n] * DIMD + d];
        const float r = residual[i] - q;
        residual[i] = r;
        res_bf[i]   = f2bf(r);
        qout[i]    += q;
        s += r * r;
    }
    #pragma unroll
    for (int m = 16; m >= 1; m >>= 1) s += __shfl_xor(s, m, 32);
    if ((threadIdx.x & 31) == 0) atomicAdd(lacc, s);
}

__global__ __launch_bounds__(256) void writeidx_kernel(
    const int* __restrict__ idx, float* __restrict__ out)
{
    int i = blockIdx.x * blockDim.x + threadIdx.x;
    if (i < NROW) out[i] = (float)idx[i];
}

__global__ void finalize_kernel(const float* __restrict__ lacc,
                                float* __restrict__ loss_out)
{
    if (threadIdx.x < NQ)
        loss_out[threadIdx.x] = lacc[threadIdx.x] * (1.0f / (float)NELEM);
}

// ---------------------------------------------------------------------------
extern "C" void kernel_launch(void* const* d_in, const int* in_sizes, int n_in,
                              void* d_out, int out_size, void* d_ws, size_t ws_size,
                              hipStream_t stream)
{
    (void)in_sizes; (void)n_in; (void)out_size; (void)ws_size;
    const float* x  = (const float*)d_in[0];                 // [8,4096,256]
    const float* cb = (const float*)d_in[1];                 // [4,4096,256]

    float* out       = (float*)d_out;
    float* qout      = out;                                  // 8388608 floats
    float* idx_out_f = out + (size_t)NELEM;                  // 4*32768 floats
    float* loss_out  = out + (size_t)NELEM + (size_t)NQ * NROW;  // 4 floats

    char* ws = (char*)d_ws;
    size_t off = 0;
    float*          residual = (float*)(ws + off);          off += (size_t)NELEM * 4;
    unsigned short* res_bf   = (unsigned short*)(ws + off); off += (size_t)NELEM * 2;
    unsigned short* cb_bf    = (unsigned short*)(ws + off); off += (size_t)NQ * KCB * DIMD * 2;
    float*          cnorm    = (float*)(ws + off);          off += (size_t)NQ * KCB * 4;
    int*            idxbuf   = (int*)(ws + off);            off += (size_t)NROW * 4;
    float*          lacc     = (float*)(ws + off);          off += (size_t)NQ * 4;

    init_kernel  <<<2048, 256, 0, stream>>>(x, residual, res_bf, qout, lacc);
    cbconv_kernel<<<1024, 256, 0, stream>>>(cb, cb_bf);
    cnorm_kernel <<<(NQ * KCB) / 8, 256, 0, stream>>>(cb, cnorm);

    for (int qi = 0; qi < NQ; ++qi) {
        argmin_kernel<<<NROW / 256, 256, 0, stream>>>(
            res_bf, cb_bf + (size_t)qi * KCB * DIMD, cnorm + (size_t)qi * KCB, idxbuf);
        update_kernel<<<2048, 256, 0, stream>>>(
            residual, res_bf, cb + (size_t)qi * KCB * DIMD, idxbuf, qout, lacc + qi);
        writeidx_kernel<<<NROW / 256, 256, 0, stream>>>(
            idxbuf, idx_out_f + (size_t)qi * NROW);
    }
    finalize_kernel<<<1, 32, 0, stream>>>(lacc, loss_out);
}
